// NegativeLogLikelihoodLoss_33045478375642
// MI455X (gfx1250) — compile-verified, matrix operand order checked by reading, not run on hardware
//
#include <hip/hip_runtime.h>

#define NDIM 8192
#define EPSV 1e-5f
#define CHUNK_W 1024   // columns per degree-wave

typedef __attribute__((ext_vector_type(16))) __bf16 v16bf;
typedef __attribute__((ext_vector_type(8)))  float  v8f;

// ---------------------------------------------------------------- zero deg
__global__ void zero_deg_kernel(float* __restrict__ deg) {
    int i = blockIdx.x * blockDim.x + threadIdx.x;
    if (i < NDIM) deg[i] = 0.0f;
}

// ---------------------------------------------------------------- Pass 1:
// deg = rowsum(graph) via v_wmma_f32_16x16x32_bf16 (D = A * ones + C).
// grid = (NDIM/16) strips * (NDIM/CHUNK_W) chunks, block = 1 wave (32).
__global__ __launch_bounds__(32)
void degree_wmma_kernel(const float* __restrict__ g, float* __restrict__ deg) {
    const int NCHUNKS = NDIM / CHUNK_W;
    const int strip = blockIdx.x / NCHUNKS;
    const int chunk = blockIdx.x % NCHUNKS;
    const int lane  = threadIdx.x;

    // 16-bit A-matrix 16x32 layout: lanes 0-15 -> rows 0..15, K {0..7,16..23};
    // lanes 16-31 -> rows 0..15, K {8..15,24..31}.
    const int row    = strip * 16 + (lane & 15);
    const int colsel = (lane >> 4) ? 8 : 0;
    const float* __restrict__ rowp = g + (size_t)row * NDIM;

    v16bf ones;
#pragma unroll
    for (int e = 0; e < 16; ++e) ones[e] = (__bf16)1.0f;

    v8f acc = {0.f, 0.f, 0.f, 0.f, 0.f, 0.f, 0.f, 0.f};

    const int kbeg = chunk * CHUNK_W;
    const int kend = kbeg + CHUNK_W;
    for (int k0 = kbeg; k0 < kend; k0 += 32) {
        const float4* p0 = (const float4*)(rowp + k0 + colsel);
        const float4* p1 = (const float4*)(rowp + k0 + colsel + 16);
        float4 f0 = p0[0];
        float4 f1 = p0[1];
        float4 f2 = p1[0];
        float4 f3 = p1[1];
        float fv[16] = { f0.x, f0.y, f0.z, f0.w,  f1.x, f1.y, f1.z, f1.w,
                         f2.x, f2.y, f2.z, f2.w,  f3.x, f3.y, f3.z, f3.w };
        v16bf a;
#pragma unroll
        for (int e = 0; e < 16; ++e) a[e] = (__bf16)fv[e];   // 0/1 exact in bf16

        acc = __builtin_amdgcn_wmma_f32_16x16x32_bf16(
            /*neg_a=*/false, a, /*neg_b=*/false, ones,
            /*c_mod=*/(short)0, acc, /*reuse_a=*/false, /*reuse_b=*/false);
    }

    // D: VGPR r, lanes 0-15 => M=r ; lanes 16-31 => M=8+r (replicated over N).
    if ((lane & 15) == 0) {
        const int mbase = strip * 16 + (lane >> 4) * 8;
#pragma unroll
        for (int r = 0; r < 8; ++r)
            atomicAdd(&deg[mbase + r], acc[r]);   // integer-valued -> deterministic
    }
}

// ---------------------------------------------------------------- Pass 2:
// per-row upper-triangle log-likelihood; block b == row i; partial per row.
__global__ __launch_bounds__(256)
void loss_rows_kernel(const float* __restrict__ g, const float* __restrict__ deg,
                      const float* __restrict__ params, float* __restrict__ partials) {
    const int i = blockIdx.x;
    const float alpha = params[0];
    const float beta  = params[1];
    const float sigma = params[2];
    const float srow  = fmaf(alpha, deg[i], sigma);
    const float* __restrict__ rowp = g + (size_t)i * NDIM;

    const int j0 = i & ~3;            // aligned start covering diagonal
    float lsum = 0.0f;

    for (int j = j0 + 4 * (int)threadIdx.x; j < NDIM; j += 4 * 256) {
        float4 g4 = *(const float4*)(rowp + j);
        float4 d4 = *(const float4*)(deg + j);
        float gv[4] = { g4.x, g4.y, g4.z, g4.w };
        float dv[4] = { d4.x, d4.y, d4.z, d4.w };
#pragma unroll
        for (int u = 0; u < 4; ++u) {
            if (j + u >= i) {
                float s = fmaf(beta, dv[u], srow);
                float e = __expf(s);                     // v_exp_f32
                float r = __builtin_amdgcn_rcpf(1.0f + e); // v_rcp_f32
                float p = r;                             // sigmoid(-s)
                float q = e * r;                         // 1 - p, accurately
                float arg = (gv[u] == 1.0f) ? (p + EPSV) : (q + EPSV);
                lsum += __logf(arg);                     // v_log_f32
            }
        }
    }

    __shared__ float red[256];
    red[threadIdx.x] = lsum;
    __syncthreads();
#pragma unroll
    for (int off = 128; off > 0; off >>= 1) {
        if ((int)threadIdx.x < off) red[threadIdx.x] += red[threadIdx.x + off];
        __syncthreads();
    }
    if (threadIdx.x == 0) partials[i] = red[0];
}

// ---------------------------------------------------------------- final
__global__ __launch_bounds__(256)
void final_reduce_kernel(const float* __restrict__ partials, float* __restrict__ out) {
    __shared__ float red[256];
    float s = 0.0f;
    for (int i = threadIdx.x; i < NDIM; i += 256) s += partials[i];  // fixed order
    red[threadIdx.x] = s;
    __syncthreads();
#pragma unroll
    for (int off = 128; off > 0; off >>= 1) {
        if ((int)threadIdx.x < off) red[threadIdx.x] += red[threadIdx.x + off];
        __syncthreads();
    }
    if (threadIdx.x == 0) out[0] = -red[0];
}

// ---------------------------------------------------------------- launch
extern "C" void kernel_launch(void* const* d_in, const int* in_sizes, int n_in,
                              void* d_out, int out_size, void* d_ws, size_t ws_size,
                              hipStream_t stream) {
    const float* params = (const float*)d_in[0];   // [3]: alpha, beta, sigma
    const float* graph  = (const float*)d_in[1];   // [8192*8192]
    float* deg      = (float*)d_ws;                // 8192 floats
    float* partials = deg + NDIM;                  // 8192 floats
    float* out      = (float*)d_out;

    zero_deg_kernel<<<NDIM / 256, 256, 0, stream>>>(deg);
    degree_wmma_kernel<<<(NDIM / 16) * (NDIM / CHUNK_W), 32, 0, stream>>>(graph, deg);
    loss_rows_kernel<<<NDIM, 256, 0, stream>>>(graph, deg, params, partials);
    final_reduce_kernel<<<1, 256, 0, stream>>>(partials, out);
}
